// GAT_80058190398139
// MI455X (gfx1250) — matrix operand order repaired and checked by reading, not compiled, not run on hardware
//
#include <hip/hip_runtime.h>
#include <hip/hip_bf16.h>
#include <math.h>

typedef __attribute__((ext_vector_type(16))) _Float16 v16h;
typedef __attribute__((ext_vector_type(8)))  float    v8f;

#define H 32
#define P 20
#define NH 4
#define HD 8
#define HL 7
#define NEG_GAT 0.2f
#define NEG_MLP 0.01f

// ---------------------------------------------------------------------------
// Node linear transforms via WMMA: xl = x@Wl^T + bl, xr = x@Wr^T + br
// One wave computes a 16-row (node) x 32-col tile for BOTH Wl and Wr,
// sharing the A fragment across 4 v_wmma_f32_16x16x32_f16 per K-chunk.
// FIN is a template constant (50 or 32) so K-bounds fold at compile time.
// Out-of-range rows are CLAMPED (their D rows are never stored), and K-pad
// uses clamped-index loads + value selects => branchless, unconditional
// global_load_b64 streams instead of exec-guarded scalar loads.
// ---------------------------------------------------------------------------
template <int FIN>
__global__ __launch_bounds__(256) void node_linear2_wmma(
    const float* __restrict__ x, int n,
    const float* __restrict__ Wl, const float* __restrict__ bl,
    const float* __restrict__ Wr, const float* __restrict__ br,
    float* __restrict__ xl, float* __restrict__ xr)
{
  int wave = (int)((blockIdx.x * blockDim.x + threadIdx.x) >> 5);
  int lane = threadIdx.x & 31;
  int row0 = wave * 16;
  if (row0 >= n) return;                 // wave-uniform: EXEC stays full

  const int lhalf = lane >> 4;           // 0 or 1
  const int lmod  = lane & 15;
  int row = row0 + lmod;
  row = (row < n) ? row : (n - 1);       // clamp: garbage rows are not stored

  const float* __restrict__ xrow = x  + (size_t)row * FIN;
  const float* __restrict__ wl0r = Wl + (size_t)lmod        * FIN;
  const float* __restrict__ wl1r = Wl + (size_t)(16 + lmod) * FIN;
  const float* __restrict__ wr0r = Wr + (size_t)lmod        * FIN;
  const float* __restrict__ wr1r = Wr + (size_t)(16 + lmod) * FIN;

  v8f accl0 = {}, accl1 = {}, accr0 = {}, accr1 = {};

#pragma unroll
  for (int k0 = 0; k0 < FIN; k0 += 32) {
    // ---- A fragment (16x32 f16): lanes 0-15 hold K {0..15}, lanes 16-31 {16..31}
    v16h a;
#pragma unroll
    for (int jj = 0; jj < 4; ++jj) {                 // first 8 K-values, as float2 pairs
      int k = k0 + lhalf * 16 + jj * 2;
      bool ok = (k + 1) < FIN;                       // FIN even => pair-uniform validity
      int kc = ok ? k : 0;
      float2 v = *(const float2*)(xrow + kc);
      a[jj * 2]     = (_Float16)(ok ? v.x : 0.0f);
      a[jj * 2 + 1] = (_Float16)(ok ? v.y : 0.0f);
    }
#pragma unroll
    for (int jj = 0; jj < 4; ++jj) {                 // second 8 K-values
      int k = k0 + lhalf * 16 + 8 + jj * 2;
      bool ok = (k + 1) < FIN;
      int kc = ok ? k : 0;
      float2 v = *(const float2*)(xrow + kc);
      a[8 + jj * 2]     = (_Float16)(ok ? v.x : 0.0f);
      a[8 + jj * 2 + 1] = (_Float16)(ok ? v.y : 0.0f);
    }
    // ---- B fragments (32x16 f16): lane = column, lane-half = K-half, VGPR j = K pair 2j,2j+1
    v16h bL0, bL1, bR0, bR1;
#pragma unroll
    for (int jj = 0; jj < 8; ++jj) {
      int k = k0 + lhalf * 16 + jj * 2;
      bool ok = (k + 1) < FIN;
      int kc = ok ? k : 0;
      float2 vL0 = *(const float2*)(wl0r + kc);
      float2 vL1 = *(const float2*)(wl1r + kc);
      float2 vR0 = *(const float2*)(wr0r + kc);
      float2 vR1 = *(const float2*)(wr1r + kc);
      bL0[jj * 2]     = (_Float16)(ok ? vL0.x : 0.0f);
      bL0[jj * 2 + 1] = (_Float16)(ok ? vL0.y : 0.0f);
      bL1[jj * 2]     = (_Float16)(ok ? vL1.x : 0.0f);
      bL1[jj * 2 + 1] = (_Float16)(ok ? vL1.y : 0.0f);
      bR0[jj * 2]     = (_Float16)(ok ? vR0.x : 0.0f);
      bR0[jj * 2 + 1] = (_Float16)(ok ? vR0.y : 0.0f);
      bR1[jj * 2]     = (_Float16)(ok ? vR1.x : 0.0f);
      bR1[jj * 2 + 1] = (_Float16)(ok ? vR1.y : 0.0f);
    }
    accl0 = __builtin_amdgcn_wmma_f32_16x16x32_f16(false, a, false, bL0, (short)0, accl0, false, false);
    accl1 = __builtin_amdgcn_wmma_f32_16x16x32_f16(false, a, false, bL1, (short)0, accl1, false, false);
    accr0 = __builtin_amdgcn_wmma_f32_16x16x32_f16(false, a, false, bR0, (short)0, accr0, false, false);
    accr1 = __builtin_amdgcn_wmma_f32_16x16x32_f16(false, a, false, bR1, (short)0, accr1, false, false);
  }

  // ---- D layout: lane col = lmod (+16 for acc*1), rows = vgpr + 8*lhalf
  float bL0v = bl[lmod], bL1v = bl[16 + lmod];
  float bR0v = br[lmod], bR1v = br[16 + lmod];
#pragma unroll
  for (int j = 0; j < 8; ++j) {
    int m = row0 + lhalf * 8 + j;
    if (m < n) {
      xl[(size_t)m * H + lmod]      = accl0[j] + bL0v;
      xl[(size_t)m * H + 16 + lmod] = accl1[j] + bL1v;
      xr[(size_t)m * H + lmod]      = accr0[j] + bR0v;
      xr[(size_t)m * H + 16 + lmod] = accr1[j] + bR1v;
    }
  }
}

// ---------------------------------------------------------------------------
// Self-loop attr: deg + segment-sum of edge_attr by dst
// ---------------------------------------------------------------------------
__global__ void deg_loop_kernel(const int* __restrict__ ei, const float* __restrict__ ea,
                                float* __restrict__ deg, float* __restrict__ loopA, int E)
{
  int e = blockIdx.x * blockDim.x + threadIdx.x;
  if (e >= E) return;
  int dst = ei[E + e];
  atomicAdd(&deg[dst], 1.0f);
#pragma unroll
  for (int c = 0; c < 4; ++c) atomicAdd(&loopA[(size_t)dst * 4 + c], ea[(size_t)e * 4 + c]);
}

__global__ void loop_div_kernel(float* __restrict__ loopA, const float* __restrict__ deg, int n)
{
  int i = blockIdx.x * blockDim.x + threadIdx.x;
  if (i >= n * 4) return;
  float d = fmaxf(deg[i >> 2], 1.0f);
  loopA[i] /= d;
}

// ---------------------------------------------------------------------------
// GAT edge pass A: alpha per edge + segment max (ordered-uint float atomicMax)
// ---------------------------------------------------------------------------
__device__ __forceinline__ unsigned fkey(float f) {
  unsigned b = __float_as_uint(f);
  return (b & 0x80000000u) ? ~b : (b | 0x80000000u);
}
__device__ __forceinline__ float funkey(unsigned k) {
  unsigned b = (k & 0x80000000u) ? (k & 0x7FFFFFFFu) : ~k;
  return __uint_as_float(b);
}

__global__ void edge_alpha_kernel(const int* __restrict__ ei,
                                  const float* __restrict__ ea,
                                  const float* __restrict__ loopA,
                                  const float* __restrict__ xl,
                                  const float* __restrict__ xr,
                                  const float* __restrict__ We,
                                  const float* __restrict__ att,
                                  float* __restrict__ alpha,
                                  unsigned* __restrict__ amax,
                                  int E, int n)
{
  int e = blockIdx.x * blockDim.x + threadIdx.x;
  int total = E + n;
  if (e >= total) return;
  int src, dst;
  const float* a4;
  if (e < E) { src = ei[e]; dst = ei[E + e]; a4 = &ea[(size_t)e * 4]; }
  else       { src = dst = e - E;            a4 = &loopA[(size_t)src * 4]; }
  float e0 = a4[0], e1 = a4[1], e2 = a4[2], e3 = a4[3];
  float al = 0.0f;
#pragma unroll
  for (int f = 0; f < H; ++f) {
    float m = xl[(size_t)src * H + f] + xr[(size_t)dst * H + f]
            + We[f * 4 + 0] * e0 + We[f * 4 + 1] * e1
            + We[f * 4 + 2] * e2 + We[f * 4 + 3] * e3;
    m = (m > 0.0f) ? m : NEG_GAT * m;
    al += m * att[f];
  }
  alpha[e] = al;
  atomicMax(&amax[dst], fkey(al));
}

// Pass B: ex = exp(alpha - amax[dst]); den[dst] += ex (ex overwrites alpha)
__global__ void edge_exp_kernel(const int* __restrict__ ei,
                                float* __restrict__ alpha,
                                const unsigned* __restrict__ amax,
                                float* __restrict__ den, int E, int n)
{
  int e = blockIdx.x * blockDim.x + threadIdx.x;
  int total = E + n;
  if (e >= total) return;
  int dst = (e < E) ? ei[E + e] : (e - E);
  float ex = __expf(alpha[e] - funkey(amax[dst]));
  alpha[e] = ex;
  atomicAdd(&den[dst], ex);
}

// Pass C: out[dst] += xl[src] * (ex/den[dst]) ; one lane per (edge,feature)
__global__ void edge_aggr_kernel(const int* __restrict__ ei,
                                 const float* __restrict__ alpha,
                                 const float* __restrict__ den,
                                 const float* __restrict__ xl,
                                 float* __restrict__ acc, int E, int n)
{
  long long tid = (long long)blockIdx.x * blockDim.x + threadIdx.x;
  long long total = (long long)(E + n) * H;
  if (tid >= total) return;
  int e = (int)(tid >> 5);
  int f = (int)(tid & 31);
  int src, dst;
  if (e < E) { src = ei[e]; dst = ei[E + e]; }
  else       { src = dst = e - E; }
  float w = alpha[e] / den[dst];
  atomicAdd(&acc[(size_t)dst * H + f], xl[(size_t)src * H + f] * w);
}

// Finalize: h = acc + bias, optional relu
__global__ void finalize_kernel(const float* __restrict__ acc, const float* __restrict__ bias,
                                float* __restrict__ h, int n, int do_relu)
{
  int tid = blockIdx.x * blockDim.x + threadIdx.x;
  if (tid >= n * H) return;
  float v = acc[tid] + bias[tid & 31];
  if (do_relu) v = fmaxf(v, 0.0f);
  h[tid] = v;
}

// ---------------------------------------------------------------------------
// Cluster means over h[:-1]
// ---------------------------------------------------------------------------
__global__ void cluster_sum_kernel(const float* __restrict__ h, const int* __restrict__ part,
                                   float* __restrict__ cm, float* __restrict__ cnt, int nm1)
{
  long long tid = (long long)blockIdx.x * blockDim.x + threadIdx.x;
  if (tid >= (long long)nm1 * H) return;
  int i = (int)(tid >> 5);
  int f = (int)(tid & 31);
  int p = part[i];
  atomicAdd(&cm[p * H + f], h[(size_t)i * H + f]);
  if (f == 0) atomicAdd(&cnt[p], 1.0f);
}

__global__ void cluster_div_kernel(float* __restrict__ cm, const float* __restrict__ cnt)
{
  int tid = blockIdx.x * blockDim.x + threadIdx.x;
  if (tid >= P * H) return;
  cm[tid] /= cnt[tid >> 5];
}

// ---------------------------------------------------------------------------
// Tail: MHA(20 tokens) + 2-layer GRU(7 steps) + hist MLP + head MLP
// Single wave32 block; thread f owns feature f.
// ---------------------------------------------------------------------------
__device__ __forceinline__ float leakyf(float x, float s) { return x > 0.0f ? x : s * x; }
__device__ __forceinline__ float sigmoidf_(float x) { return 1.0f / (1.0f + __expf(-x)); }

__global__ __launch_bounds__(32) void tail_kernel(
    const float* __restrict__ cm,
    const int*   __restrict__ hist_act,
    const float* __restrict__ hist_r1,
    const float* __restrict__ diff_reward,
    const float* mha_in_w, const float* mha_in_b,
    const float* mha_out_w, const float* mha_out_b,
    const float* g0_Wih, const float* g0_Whh, const float* g0_bih, const float* g0_bhh,
    const float* g1_Wih, const float* g1_Whh, const float* g1_bih, const float* g1_bhh,
    const float* hW1, const float* hb1, const float* hW2, const float* hb2,
    const float* hW3, const float* hb3,
    const float* dW1, const float* db1, const float* dW2, const float* db2,
    const float* dW3, const float* db3,
    float* __restrict__ out)
{
  __shared__ float s_q[P][H], s_k[P][H], s_v[P][H], s_o[P][H];
  __shared__ float s_ys0[HL][H], s_h[H], s_tmp[H], s_flat[(P + 1) * H], s_hin[H + 2];
  int f = threadIdx.x;   // 0..31

  // ---- MHA qkv
  for (int t = 0; t < P; ++t) {
    float q = mha_in_b[f], k = mha_in_b[H + f], v = mha_in_b[2 * H + f];
    for (int c = 0; c < H; ++c) {
      float xc = cm[t * H + c];
      q += mha_in_w[(f)         * H + c] * xc;
      k += mha_in_w[(H + f)     * H + c] * xc;
      v += mha_in_w[(2 * H + f) * H + c] * xc;
    }
    s_q[t][f] = q; s_k[t][f] = k; s_v[t][f] = v;
  }
  __syncthreads();

  // ---- attention: 80 (head,query) tasks over 32 threads
  for (int task = f; task < NH * P; task += 32) {
    int hh = task / P, qt = task % P;
    float sc[P]; float mx = -1e30f;
    for (int kt = 0; kt < P; ++kt) {
      float s = 0.0f;
      for (int d = 0; d < HD; ++d) s += s_q[qt][hh * HD + d] * s_k[kt][hh * HD + d];
      s *= 0.35355339059327379f;  // 1/sqrt(8)
      sc[kt] = s; mx = fmaxf(mx, s);
    }
    float dn = 0.0f;
    for (int kt = 0; kt < P; ++kt) { sc[kt] = __expf(sc[kt] - mx); dn += sc[kt]; }
    for (int d = 0; d < HD; ++d) {
      float o = 0.0f;
      for (int kt = 0; kt < P; ++kt) o += sc[kt] * s_v[kt][hh * HD + d];
      s_o[qt][hh * HD + d] = o / dn;
    }
  }
  __syncthreads();

  // ---- out projection -> flat[0:640]
  for (int t = 0; t < P; ++t) {
    float o = mha_out_b[f];
    for (int c = 0; c < H; ++c) o += mha_out_w[f * H + c] * s_o[t][c];
    s_flat[t * H + f] = o;
  }
  __syncthreads();

  // ---- GRU layer 0 (one-hot input => column select of Wih)
  s_h[f] = 0.0f; __syncthreads();
  for (int t = 0; t < HL; ++t) {
    int act = hist_act[20 - HL + t];
    float gi_r = g0_Wih[(f)         * 28 + act] + g0_bih[f];
    float gi_z = g0_Wih[(H + f)     * 28 + act] + g0_bih[H + f];
    float gi_n = g0_Wih[(2 * H + f) * 28 + act] + g0_bih[2 * H + f];
    float gh_r = g0_bhh[f], gh_z = g0_bhh[H + f], gh_n = g0_bhh[2 * H + f];
    for (int c = 0; c < H; ++c) {
      float hc = s_h[c];
      gh_r += g0_Whh[(f)         * H + c] * hc;
      gh_z += g0_Whh[(H + f)     * H + c] * hc;
      gh_n += g0_Whh[(2 * H + f) * H + c] * hc;
    }
    float r = sigmoidf_(gi_r + gh_r);
    float z = sigmoidf_(gi_z + gh_z);
    float nc = tanhf(gi_n + r * gh_n);
    float hn = (1.0f - z) * nc + z * s_h[f];
    __syncthreads();
    s_h[f] = hn; s_ys0[t][f] = hn;
    __syncthreads();
  }
  // ---- GRU layer 1, ga = mean of outputs
  s_h[f] = 0.0f; float ga = 0.0f; __syncthreads();
  for (int t = 0; t < HL; ++t) {
    float gi_r = g1_bih[f], gi_z = g1_bih[H + f], gi_n = g1_bih[2 * H + f];
    float gh_r = g1_bhh[f], gh_z = g1_bhh[H + f], gh_n = g1_bhh[2 * H + f];
    for (int c = 0; c < H; ++c) {
      float xc = s_ys0[t][c], hc = s_h[c];
      gi_r += g1_Wih[(f)         * H + c] * xc;
      gi_z += g1_Wih[(H + f)     * H + c] * xc;
      gi_n += g1_Wih[(2 * H + f) * H + c] * xc;
      gh_r += g1_Whh[(f)         * H + c] * hc;
      gh_z += g1_Whh[(H + f)     * H + c] * hc;
      gh_n += g1_Whh[(2 * H + f) * H + c] * hc;
    }
    float r = sigmoidf_(gi_r + gh_r);
    float z = sigmoidf_(gi_z + gh_z);
    float nc = tanhf(gi_n + r * gh_n);
    float hn = (1.0f - z) * nc + z * s_h[f];
    __syncthreads();
    s_h[f] = hn; ga += hn;
    __syncthreads();
  }
  ga *= (1.0f / (float)HL);

  // ---- hist MLP
  float effect = 0.0f;
  for (int t = 20 - HL; t < 20; ++t) effect += hist_r1[t];
  float flag = (effect > (float)HL * 0.5f) ? 1.0f : 0.0f;
  s_hin[f] = ga;
  if (f == 0) { s_hin[H] = flag; s_hin[H + 1] = diff_reward[0]; }
  __syncthreads();
  float h1 = hb1[f];
  for (int c = 0; c < H + 2; ++c) h1 += hW1[f * (H + 2) + c] * s_hin[c];
  s_tmp[f] = leakyf(h1, NEG_MLP); __syncthreads();
  float h2 = hb2[f];
  for (int c = 0; c < H; ++c) h2 += hW2[f * H + c] * s_tmp[c];
  h2 = leakyf(h2, NEG_MLP);
  __syncthreads(); s_tmp[f] = h2; __syncthreads();
  float h3 = hb3[f];
  for (int c = 0; c < H; ++c) h3 += hW3[f * H + c] * s_tmp[c];
  s_flat[P * H + f] = h3;       // flat[640:672]
  __syncthreads();

  // ---- head MLP
  float e1 = db1[f];
  for (int c = 0; c < (P + 1) * H; ++c) e1 += dW1[f * (P + 1) * H + c] * s_flat[c];
  s_tmp[f] = leakyf(e1, NEG_MLP); __syncthreads();
  float e2 = db2[f];
  for (int c = 0; c < H; ++c) e2 += dW2[f * H + c] * s_tmp[c];
  e2 = leakyf(e2, NEG_MLP);
  __syncthreads(); s_tmp[f] = e2; __syncthreads();
  if (f < P) {
    float o = db3[f];
    for (int c = 0; c < H; ++c) o += dW3[f * H + c] * s_tmp[c];
    out[f] = o;
  }
}

// ---------------------------------------------------------------------------
// Launch
// ---------------------------------------------------------------------------
static inline unsigned ceil_div(long long a, long long b) { return (unsigned)((a + b - 1) / b); }

extern "C" void kernel_launch(void* const* d_in, const int* in_sizes, int n_in,
                              void* d_out, int out_size, void* d_ws, size_t ws_size,
                              hipStream_t stream) {
  const float* x        = (const float*)d_in[0];
  const int*   ei       = (const int*)  d_in[1];
  const float* ea       = (const float*)d_in[2];
  const int*   part     = (const int*)  d_in[3];
  const int*   hist_act = (const int*)  d_in[4];
  const float* hist_r1  = (const float*)d_in[5];
  const float* diff_r   = (const float*)d_in[6];

  const int FIN = 50;
  const int N = in_sizes[0] / FIN;
  const int E = in_sizes[1] / 2;

  // params: 7 = conv1 base, 14 = conv2, 21 = conv3, 28 = mha, 32 = gru0, 36 = gru1, 40 = hist, 46 = head
  const float* convP[3][7];
  for (int l = 0; l < 3; ++l)
    for (int j = 0; j < 7; ++j)
      convP[l][j] = (const float*)d_in[7 + l * 7 + j];
  const float* mha_in_w  = (const float*)d_in[28];
  const float* mha_in_b  = (const float*)d_in[29];
  const float* mha_out_w = (const float*)d_in[30];
  const float* mha_out_b = (const float*)d_in[31];
  const float* g0_Wih = (const float*)d_in[32];
  const float* g0_Whh = (const float*)d_in[33];
  const float* g0_bih = (const float*)d_in[34];
  const float* g0_bhh = (const float*)d_in[35];
  const float* g1_Wih = (const float*)d_in[36];
  const float* g1_Whh = (const float*)d_in[37];
  const float* g1_bih = (const float*)d_in[38];
  const float* g1_bhh = (const float*)d_in[39];
  const float* hW1 = (const float*)d_in[40];
  const float* hb1 = (const float*)d_in[41];
  const float* hW2 = (const float*)d_in[42];
  const float* hb2 = (const float*)d_in[43];
  const float* hW3 = (const float*)d_in[44];
  const float* hb3 = (const float*)d_in[45];
  const float* dW1 = (const float*)d_in[46];
  const float* db1 = (const float*)d_in[47];
  const float* dW2 = (const float*)d_in[48];
  const float* db2 = (const float*)d_in[49];
  const float* dW3 = (const float*)d_in[50];
  const float* db3 = (const float*)d_in[51];

  // workspace carve-out
  size_t off = 0;
  char* base = (char*)d_ws;
  auto carve = [&](size_t bytes) { char* p = base + off; off += (bytes + 255) & ~(size_t)255; return (void*)p; };
  float*    deg   = (float*)   carve((size_t)N * 4);
  float*    loopA = (float*)   carve((size_t)N * 4 * 4);
  float*    xl    = (float*)   carve((size_t)N * H * 4);
  float*    xr    = (float*)   carve((size_t)N * H * 4);
  float*    acc   = (float*)   carve((size_t)N * H * 4);
  float*    hfeat = (float*)   carve((size_t)N * H * 4);
  unsigned* amax  = (unsigned*)carve((size_t)N * 4);
  float*    den   = (float*)   carve((size_t)N * 4);
  float*    alpha = (float*)   carve((size_t)(E + N) * 4);
  float*    cmb   = (float*)   carve((size_t)P * H * 4);
  float*    cnt   = (float*)   carve((size_t)P * 4);
  (void)ws_size; (void)n_in; (void)out_size;

  const int T = 256;
  const int total = E + N;
  const long long waves = (N + 15) / 16;

  // self-loop edge attributes (fill_value='mean')
  hipMemsetAsync(deg,   0, (size_t)N * 4, stream);
  hipMemsetAsync(loopA, 0, (size_t)N * 4 * 4, stream);
  deg_loop_kernel<<<ceil_div(E, T), T, 0, stream>>>(ei, ea, deg, loopA, E);
  loop_div_kernel<<<ceil_div((long long)N * 4, T), T, 0, stream>>>(loopA, deg, N);

  const float* hin = x;
  for (int l = 0; l < 3; ++l) {
    const float* Wl   = convP[l][0];
    const float* bl   = convP[l][1];
    const float* Wr   = convP[l][2];
    const float* br   = convP[l][3];
    const float* We   = convP[l][4];
    const float* attp = convP[l][5];
    const float* bias = convP[l][6];

    // WMMA node transforms (FIN specialized at compile time)
    if (l == 0)
      node_linear2_wmma<50><<<ceil_div(waves * 32, T), T, 0, stream>>>(hin, N, Wl, bl, Wr, br, xl, xr);
    else
      node_linear2_wmma<32><<<ceil_div(waves * 32, T), T, 0, stream>>>(hin, N, Wl, bl, Wr, br, xl, xr);

    hipMemsetAsync(amax, 0, (size_t)N * 4, stream);          // key 0 == -inf sentinel (every dst has self-loop)
    hipMemsetAsync(den,  0, (size_t)N * 4, stream);
    hipMemsetAsync(acc,  0, (size_t)N * H * 4, stream);

    edge_alpha_kernel<<<ceil_div(total, T), T, 0, stream>>>(ei, ea, loopA, xl, xr, We, attp, alpha, amax, E, N);
    edge_exp_kernel  <<<ceil_div(total, T), T, 0, stream>>>(ei, alpha, amax, den, E, N);
    edge_aggr_kernel <<<ceil_div((long long)total * H, T), T, 0, stream>>>(ei, alpha, den, xl, acc, E, N);
    finalize_kernel  <<<ceil_div((long long)N * H, T), T, 0, stream>>>(acc, bias, hfeat, N, (l < 2) ? 1 : 0);

    hin = hfeat;
  }

  // cluster means over h[:-1]
  hipMemsetAsync(cmb, 0, (size_t)P * H * 4, stream);
  hipMemsetAsync(cnt, 0, (size_t)P * 4, stream);
  cluster_sum_kernel<<<ceil_div((long long)(N - 1) * H, T), T, 0, stream>>>(hfeat, part, cmb, cnt, N - 1);
  cluster_div_kernel<<<1, P * H, 0, stream>>>(cmb, cnt);

  tail_kernel<<<1, 32, 0, stream>>>(
      cmb, hist_act, hist_r1, diff_r,
      mha_in_w, mha_in_b, mha_out_w, mha_out_b,
      g0_Wih, g0_Whh, g0_bih, g0_bhh,
      g1_Wih, g1_Whh, g1_bih, g1_bhh,
      hW1, hb1, hW2, hb2, hW3, hb3,
      dW1, db1, dW2, db2, dW3, db3,
      (float*)d_out);
}
